// SingleBandFlow_89816356094034
// MI455X (gfx1250) — matrix operand order repaired and checked by reading, not compiled, not run on hardware
//
#include <hip/hip_runtime.h>
#include <math.h>

// ---------------------------------------------------------------------------
// Problem constants (match reference)
// ---------------------------------------------------------------------------
#define BATCH      16384
#define NIN        9
#define NHID       128
#define NLAYERS    10
#define CTXD       64
#define INCH       4
#define SEQ        256
#define BN_EPS     1e-5f
#define LOG2PI_F   1.8378770664093453f

#define K1PAD      96     // 73 -> 96 (3 chunks of 32)
#define N3PAD      32     // 18 -> 32 (2 tiles of 16)
#define RBK        96     // rb conv GEMM K = 3 taps * 32 ch
#define RBW_ELEMS  (2 * 32 * RBK)          // both rb conv weight blocks, f16
#define RBW_BYTES  (RBW_ELEMS * 2)         // 12288 bytes

typedef __attribute__((ext_vector_type(16))) _Float16 v16h;
typedef __attribute__((ext_vector_type(8)))  _Float16 v8h;
typedef __attribute__((ext_vector_type(8)))  float    v8f;
typedef __attribute__((ext_vector_type(4)))  unsigned v4u;
typedef __attribute__((ext_vector_type(8)))  int      v8i;
typedef __attribute__((ext_vector_type(4)))  int      v4i;

// TDM availability (device pass only; host pass parses the fallback).
#if defined(__HIP_DEVICE_COMPILE__) && defined(__has_builtin)
#  if __has_builtin(__builtin_amdgcn_tensor_load_to_lds) && \
      __has_builtin(__builtin_amdgcn_s_wait_tensorcnt)
#    define USE_TDM 1
#  endif
#endif
#ifndef USE_TDM
#  define USE_TDM 0
#endif

// ---------------------------------------------------------------------------
// WMMA fragment loaders (wave32, V_WMMA_F32_16X16X32_F16)
//
// A (16x32 f16), per CDNA5 ISA 7.12.2:
//   lane m = lane&15 owns row M=m.  kb = half*8 (half = lane>>4).
//   v16h elements 0..7  hold K = k0*32 + kb + 0..7
//   v16h elements 8..15 hold K = k0*32 + 16 + kb + 0..7
// B (32x16 f16): lane owns column n; same K order as A (kb = half*8,
//   elements 0..7 K=k0*32+kb+0..7, elements 8..15 K=k0*32+16+kb+0..7).
// C/D (16x16 f32): element r holds M = r + half*8, N = lane&15.
// ---------------------------------------------------------------------------
__device__ __forceinline__ v16h load_fragA(const _Float16* base, int stride,
                                           int m, int half, int k0) {
  const _Float16* p = base + m * stride + k0 * 32 + half * 8;
  v8h lo = *(const v8h*)(p);
  v8h hi = *(const v8h*)(p + 16);
  return __builtin_shufflevector(lo, hi, 0,1,2,3,4,5,6,7,8,9,10,11,12,13,14,15);
}

// Conv im2col A-fragment: rowp = &inT[t + k0][0]; row is 32 contiguous f16
// (channel index == K index within the chunk).
__device__ __forceinline__ v16h load_fragA_conv(const _Float16* rowp, int half) {
  v8h lo = *(const v8h*)(rowp + half * 8);
  v8h hi = *(const v8h*)(rowp + 16 + half * 8);
  return __builtin_shufflevector(lo, hi, 0,1,2,3,4,5,6,7,8,9,10,11,12,13,14,15);
}

__device__ __forceinline__ v16h load_fragB(const _Float16* wt, int stride,
                                           int n, int half, int k0) {
  const _Float16* p = wt + n * stride + k0 * 32 + half * 8;
  v8h lo = *(const v8h*)(p);
  v8h hi = *(const v8h*)(p + 16);
  return __builtin_shufflevector(lo, hi, 0,1,2,3,4,5,6,7,8,9,10,11,12,13,14,15);
}

#define WMMA_F16(a, b, c) \
  __builtin_amdgcn_wmma_f32_16x16x32_f16(false, (a), false, (b), (short)0, (c), false, false)

// ---------------------------------------------------------------------------
// Weight prep kernels (f32 -> f16, transposed / padded layouts)
// ---------------------------------------------------------------------------
__global__ void prep_w1(const float* __restrict__ W1, _Float16* __restrict__ W1t) {
  int idx = blockIdx.x * 256 + threadIdx.x;            // [10][128][96]
  if (idx >= NLAYERS * NHID * K1PAD) return;
  int k = idx % K1PAD;
  int n = (idx / K1PAD) % NHID;
  int l = idx / (K1PAD * NHID);
  float v = (k < NIN + CTXD) ? W1[(l * (NIN + CTXD) + k) * NHID + n] : 0.f;
  W1t[idx] = (_Float16)v;
}

__global__ void prep_w2(const float* __restrict__ W2, _Float16* __restrict__ W2t) {
  int idx = blockIdx.x * 256 + threadIdx.x;            // [10][128][128]
  if (idx >= NLAYERS * NHID * NHID) return;
  int k = idx % NHID;
  int n = (idx / NHID) % NHID;
  int l = idx / (NHID * NHID);
  W2t[idx] = (_Float16)W2[(l * NHID + k) * NHID + n];
}

__global__ void prep_w3(const float* __restrict__ W3, _Float16* __restrict__ W3t) {
  int idx = blockIdx.x * 256 + threadIdx.x;            // [10][32][128]
  if (idx >= NLAYERS * N3PAD * NHID) return;
  int k = idx % NHID;
  int n = (idx / NHID) % N3PAD;
  int l = idx / (NHID * N3PAD);
  float v = (n < 2 * NIN) ? W3[(l * NHID + k) * (2 * NIN) + n] : 0.f;
  W3t[idx] = (_Float16)v;
}

// rb conv weights -> B[c][k], k = dt*32 + ci.  [2][32][96] f16 (contiguous)
__global__ void prep_rbw(const float* __restrict__ rw1, const float* __restrict__ rw2,
                         _Float16* __restrict__ rbB) {
  int idx = blockIdx.x * 256 + threadIdx.x;
  if (idx >= RBW_ELEMS) return;
  int k    = idx % RBK;
  int c    = (idx / RBK) % 32;
  int conv = idx / (RBK * 32);
  int dt = k / 32, ci = k % 32;
  const float* w = conv ? rw2 : rw1;
  rbB[idx] = (_Float16)w[(c * 32 + ci) * 3 + dt];
}

// ---------------------------------------------------------------------------
// Context encoder, 4 recompute passes, one block (256 thr = 8 waves) / sample.
// conv1 on VALU (K=20 only); both resblock convs on WMMA via the transposed
// im2col layout.  rb conv B-weights staged into LDS once per block via the
// Tensor Data Mover (TENSOR_LOAD_TO_LDS + s_wait_tensorcnt), overlapped with
// the curve and conv1-weight staging.  bnp layout: [bn_idx][ {a[32], c[32]} ].
//   stage 1: conv1 stats   stage 2: rb1 stats
//   stage 3: rb2 stats     stage 4: final -> context[B,64]
// ---------------------------------------------------------------------------
__global__ __launch_bounds__(256)
void enc_pass(int stage,
              const float* __restrict__ curve,
              const float* __restrict__ conv1_w, const float* __restrict__ conv1_b,
              const float* __restrict__ rb_b1,   const float* __restrict__ rb_b2,
              const _Float16* __restrict__ rbB,
              const float* __restrict__ lin_w,   const float* __restrict__ lin_b,
              const float* __restrict__ bnp,
              float* __restrict__ partials, float* __restrict__ partials2,
              float* __restrict__ ctx) {
  const int s    = blockIdx.x;
  const int tid  = threadIdx.x;
  const int wave = tid >> 5;
  const int lane = tid & 31;
  const int m    = lane & 15;
  const int half = lane >> 4;

  __shared__ float sC[INCH][SEQ + 4];                               // halo 2
  __shared__ __attribute__((aligned(16))) _Float16 sXT[SEQ + 2][32]; // x^T, halo 1
  __shared__ __attribute__((aligned(16))) _Float16 sHT[SEQ + 2][32]; // h1r^T
  __shared__ __attribute__((aligned(16))) _Float16 sB[2][32][RBK];   // rb B-weights
  __shared__ float sW[2048];
  __shared__ float sRed[256];
  __shared__ float sStat[4][8][32];
  __shared__ float sPool[32];

#if USE_TDM
  // ---- TDM: DMA both rb weight blocks (12288 B) global -> LDS, wave 0 ----
  if (tid < 32) {
    unsigned lds_addr = (unsigned)(size_t)&sB[0][0][0];  // low 32 bits = LDS offset
    unsigned long long ga = (unsigned long long)(size_t)rbB;
    // D# group 0: count=1 | lds_addr | global_addr[56:0] | type=2
    v4u g0;
    g0.x = 1u;                                   // count=1, user descriptor
    g0.y = lds_addr;
    g0.z = (unsigned)(ga & 0xFFFFFFFFu);
    g0.w = (unsigned)((ga >> 32) & 0x01FFFFFFu) | (2u << 30);
    // D# group 1: data_size=1 (2B), tensor_dim0=tile_dim0=6144, dim1=1,
    //             tensor_dim0_stride=6144, everything else 0.
    v8i g1;
    g1[0] = (int)(1u << 16);                        // data_size = 2 bytes
    g1[1] = (int)((RBW_ELEMS & 0xFFFFu) << 16);     // tensor_dim0[15:0]
    g1[2] = (int)((RBW_ELEMS >> 16) | (1u << 16));  // tensor_dim0[31:16] | tensor_dim1[15:0]=1
    g1[3] = (int)((unsigned)RBW_ELEMS << 16);       // tile_dim0 = 6144
    g1[4] = 1;                                      // tile_dim1 = 1, tile_dim2 = 0
    g1[5] = RBW_ELEMS;                              // tensor_dim0_stride[31:0]
    g1[6] = 0;
    g1[7] = 0;
    v4i g2 = {0, 0, 0, 0};
    v4i g3 = {0, 0, 0, 0};
#if defined(__clang_major__) && (__clang_major__ >= 23)
    v8i g4 = {0, 0, 0, 0, 0, 0, 0, 0};
    __builtin_amdgcn_tensor_load_to_lds(g0, g1, g2, g3, g4, 0);
#else
    __builtin_amdgcn_tensor_load_to_lds(g0, g1, g2, g3, 0);
#endif
  }
#else
  for (int i = tid; i < RBW_ELEMS; i += 256) (&sB[0][0][0])[i] = rbB[i];
#endif

  // curve with zero halo (overlaps with the TDM transfer)
  for (int i = tid; i < INCH * (SEQ + 4); i += 256) {
    int ci = i / (SEQ + 4), t = i % (SEQ + 4);
    sC[ci][t] = (t >= 2 && t < SEQ + 2) ? curve[(s * INCH + ci) * SEQ + (t - 2)] : 0.f;
  }
  if (tid < 32) {
    sXT[0][tid] = (_Float16)0.f; sXT[SEQ + 1][tid] = (_Float16)0.f;
    sHT[0][tid] = (_Float16)0.f; sHT[SEQ + 1][tid] = (_Float16)0.f;
  }
  for (int i = tid; i < 32 * INCH * 5; i += 256) sW[i] = conv1_w[i];

#if USE_TDM
  if (tid < 32) __builtin_amdgcn_s_wait_tensorcnt(0);
#endif
  __syncthreads();

  // ---- conv1 (k=5, SAME) on VALU; thread = (channel c, t-group tg) ----
  {
    const int c  = tid >> 3;
    const int tg = tid & 7;
    const float b0 = conv1_b[c];
    const float a0 = bnp[0 * 64 + c];
    const float c0 = bnp[0 * 64 + 32 + c];
    float sum = 0.f, sq = 0.f;
    for (int it = 0; it < 32; ++it) {
      int t = tg * 32 + it;
      float acc = b0;
      #pragma unroll
      for (int ci = 0; ci < INCH; ++ci)
        #pragma unroll
        for (int dk = 0; dk < 5; ++dk)
          acc += sW[(c * INCH + ci) * 5 + dk] * sC[ci][t + dk];
      if (stage == 1) { sum += acc; sq += acc * acc; }
      else { float x = a0 * acc + c0; x = x > 0.f ? x : 0.f;
             sXT[t + 1][c] = (_Float16)x; }
    }
    if (stage == 1) {
      sRed[tid] = sum; __syncthreads();
      if (tg == 0) { float v = 0.f; for (int j = 0; j < 8; ++j) v += sRed[c * 8 + j];
                     partials[blockIdx.x * 32 + c] = v; }
      __syncthreads();
      sRed[tid] = sq; __syncthreads();
      if (tg == 0) { float v = 0.f; for (int j = 0; j < 8; ++j) v += sRed[c * 8 + j];
                     partials2[blockIdx.x * 32 + c] = v; }
      return;
    }
  }
  __syncthreads();

  const _Float16* sB1 = &sB[0][0][0];
  const _Float16* sB2 = &sB[1][0][0];

  // ---- resblock conv1 on WMMA: h1 = x (*) rb_w1;  16 M-tiles over 8 waves ----
  {
    float sum0 = 0.f, sq0 = 0.f, sum1 = 0.f, sq1 = 0.f;
    #pragma unroll
    for (int it = 0; it < 2; ++it) {
      const int mt = wave * 2 + it;
      #pragma unroll
      for (int n0 = 0; n0 < 2; ++n0) {
        v8f acc = {0.f, 0.f, 0.f, 0.f, 0.f, 0.f, 0.f, 0.f};
        #pragma unroll
        for (int k0 = 0; k0 < 3; ++k0) {   // tap dt == k0
          v16h af = load_fragA_conv(&sXT[mt * 16 + m + k0][0], half);
          v16h bf = load_fragB(sB1, RBK, n0 * 16 + m, half, k0);
          acc = WMMA_F16(af, bf, acc);
        }
        const int col = n0 * 16 + m;
        const float bb = rb_b1[col];
        if (stage == 2) {
          float sv = 0.f, qv = 0.f;
          #pragma unroll
          for (int r = 0; r < 8; ++r) { float h = acc[r] + bb; sv += h; qv += h * h; }
          if (n0 == 0) { sum0 += sv; sq0 += qv; } else { sum1 += sv; sq1 += qv; }
        } else {
          const float a1 = bnp[1 * 64 + col];
          const float c1 = bnp[1 * 64 + 32 + col];
          #pragma unroll
          for (int r = 0; r < 8; ++r) {
            float h = a1 * (acc[r] + bb) + c1; h = h > 0.f ? h : 0.f;
            sHT[mt * 16 + r + half * 8 + 1][col] = (_Float16)h;
          }
        }
      }
    }
    if (stage == 2) {
      sStat[0][wave][lane] = sum0; sStat[1][wave][lane] = sq0;
      sStat[2][wave][lane] = sum1; sStat[3][wave][lane] = sq1;
      __syncthreads();
      if (tid < 32) {
        const int c = tid, cc = c & 15;
        const int sS = (c < 16) ? 0 : 2, sQ = sS + 1;
        float sv = 0.f, qv = 0.f;
        #pragma unroll
        for (int w2 = 0; w2 < 8; ++w2) {
          sv += sStat[sS][w2][cc] + sStat[sS][w2][16 + cc];
          qv += sStat[sQ][w2][cc] + sStat[sQ][w2][16 + cc];
        }
        partials [blockIdx.x * 32 + c] = sv;
        partials2[blockIdx.x * 32 + c] = qv;
      }
      return;
    }
  }
  __syncthreads();

  // ---- resblock conv2 on WMMA: h2 = h1r (*) rb_w2 ----
  {
    float sum0 = 0.f, sq0 = 0.f, sum1 = 0.f, sq1 = 0.f;  // stage 3: stats; stage 4: pool
    #pragma unroll
    for (int it = 0; it < 2; ++it) {
      const int mt = wave * 2 + it;
      #pragma unroll
      for (int n0 = 0; n0 < 2; ++n0) {
        v8f acc = {0.f, 0.f, 0.f, 0.f, 0.f, 0.f, 0.f, 0.f};
        #pragma unroll
        for (int k0 = 0; k0 < 3; ++k0) {
          v16h af = load_fragA_conv(&sHT[mt * 16 + m + k0][0], half);
          v16h bf = load_fragB(sB2, RBK, n0 * 16 + m, half, k0);
          acc = WMMA_F16(af, bf, acc);
        }
        const int col = n0 * 16 + m;
        const float bb = rb_b2[col];
        if (stage == 3) {
          float sv = 0.f, qv = 0.f;
          #pragma unroll
          for (int r = 0; r < 8; ++r) { float h = acc[r] + bb; sv += h; qv += h * h; }
          if (n0 == 0) { sum0 += sv; sq0 += qv; } else { sum1 += sv; sq1 += qv; }
        } else {
          const float a2 = bnp[2 * 64 + col];
          const float c2 = bnp[2 * 64 + 32 + col];
          float pv = 0.f;
          #pragma unroll
          for (int r = 0; r < 8; ++r) {
            const int row = mt * 16 + r + half * 8 + 1;
            float y = a2 * (acc[r] + bb) + c2 + (float)sXT[row][col];  // residual
            y = y > 0.f ? y : 0.f; pv += y;
          }
          if (n0 == 0) sum0 += pv; else sum1 += pv;
        }
      }
    }
    sStat[0][wave][lane] = sum0; sStat[1][wave][lane] = sq0;
    sStat[2][wave][lane] = sum1; sStat[3][wave][lane] = sq1;
    __syncthreads();
    if (stage == 3) {
      if (tid < 32) {
        const int c = tid, cc = c & 15;
        const int sS = (c < 16) ? 0 : 2, sQ = sS + 1;
        float sv = 0.f, qv = 0.f;
        #pragma unroll
        for (int w2 = 0; w2 < 8; ++w2) {
          sv += sStat[sS][w2][cc] + sStat[sS][w2][16 + cc];
          qv += sStat[sQ][w2][cc] + sStat[sQ][w2][16 + cc];
        }
        partials [blockIdx.x * 32 + c] = sv;
        partials2[blockIdx.x * 32 + c] = qv;
      }
      return;
    }
    // stage 4: pooled mean per channel
    if (tid < 32) {
      const int c = tid, cc = c & 15;
      const int sS = (c < 16) ? 0 : 2;
      float sv = 0.f;
      #pragma unroll
      for (int w2 = 0; w2 < 8; ++w2)
        sv += sStat[sS][w2][cc] + sStat[sS][w2][16 + cc];
      sPool[c] = sv * (1.0f / SEQ);
    }
  }
  __syncthreads();

  // ---- stage 4: linear(32->64) + tanh ----
  for (int i = tid; i < 32 * CTXD; i += 256) sW[i] = lin_w[i];
  __syncthreads();
  if (tid < CTXD) {
    float acc = lin_b[tid];
    #pragma unroll
    for (int ci = 0; ci < 32; ++ci) acc += sPool[ci] * sW[ci * CTXD + tid];
    ctx[s * CTXD + tid] = tanhf(acc);
  }
}

// One block per channel; deterministic tree reduction of per-block partials.
__global__ __launch_bounds__(256)
void bn_finalize(const float* __restrict__ partials, const float* __restrict__ partials2,
                 const float* __restrict__ g, const float* __restrict__ b,
                 float* __restrict__ bnp_out) {
  const int c = blockIdx.x, tid = threadIdx.x;
  __shared__ float r1[256], r2[256];
  float s = 0.f, q = 0.f;
  for (int i = tid; i < BATCH; i += 256) {
    s += partials[i * 32 + c];
    q += partials2[i * 32 + c];
  }
  r1[tid] = s; r2[tid] = q; __syncthreads();
  for (int off = 128; off > 0; off >>= 1) {
    if (tid < off) { r1[tid] += r1[tid + off]; r2[tid] += r2[tid + off]; }
    __syncthreads();
  }
  if (tid == 0) {
    const float N = (float)BATCH * (float)SEQ;
    float mean = r1[0] / N;
    float var  = r2[0] / N - mean * mean;
    float a = g[c] * rsqrtf(var + BN_EPS);
    bnp_out[c]      = a;
    bnp_out[32 + c] = b[c] - mean * a;
  }
}

// ---------------------------------------------------------------------------
// Coupling flow: 4 waves/block, each wave owns 16 batch rows through all
// 10 layers; all three matmuls per layer on v_wmma_f32_16x16x32_f16.
// ---------------------------------------------------------------------------
__global__ __launch_bounds__(128)
void coupling(const float* __restrict__ inputs, const float* __restrict__ ctx,
              const _Float16* __restrict__ W1t, const _Float16* __restrict__ W2t,
              const _Float16* __restrict__ W3t,
              const float* __restrict__ b1, const float* __restrict__ b2,
              const float* __restrict__ b3, float* __restrict__ out) {
  __shared__ __attribute__((aligned(16))) _Float16 sA [4][16][K1PAD];
  __shared__ __attribute__((aligned(16))) _Float16 sH1[4][16][NHID];
  __shared__ __attribute__((aligned(16))) _Float16 sH2[4][16][NHID];
  __shared__ float sOut[4][16][N3PAD];

  const int tid  = threadIdx.x;
  const int wave = tid >> 5;
  const int lane = tid & 31;
  const int m    = lane & 15;
  const int half = lane >> 4;
  const int rowbase = blockIdx.x * 64 + wave * 16;

  {
    const float* crow = ctx + (size_t)(rowbase + m) * CTXD;
    for (int j = 0; j < 32; ++j)
      sA[wave][m][NIN + half * 32 + j] = (_Float16)crow[half * 32 + j];
    if (half == 0)
      for (int k = NIN + CTXD; k < K1PAD; ++k) sA[wave][m][k] = (_Float16)0.f;
  }

  float z[NIN];
  float ld = 0.f;
  if (half == 0) {
    const float* zin = inputs + (size_t)(rowbase + m) * NIN;
    #pragma unroll
    for (int i = 0; i < NIN; ++i) z[i] = zin[i];
  }

  for (int l = 0; l < NLAYERS; ++l) {
    if (l + 1 < NLAYERS)
      __builtin_prefetch(W2t + (size_t)(l + 1) * NHID * NHID, 0, 1);

    if (half == 0) {
      #pragma unroll
      for (int i = 0; i < NIN; ++i) {
        bool masked = ((i & 1) == (l & 1));
        sA[wave][m][i] = (_Float16)(masked ? z[i] : 0.f);
      }
    }
    __syncthreads();

    const _Float16* w1 = W1t + (size_t)l * NHID * K1PAD;
    const _Float16* w2 = W2t + (size_t)l * NHID * NHID;
    const _Float16* w3 = W3t + (size_t)l * N3PAD * NHID;
    const float* bb1 = b1 + l * NHID;
    const float* bb2 = b2 + l * NHID;
    const float* bb3 = b3 + l * (2 * NIN);

    #pragma unroll
    for (int n0 = 0; n0 < 8; ++n0) {
      v8f acc = {0.f, 0.f, 0.f, 0.f, 0.f, 0.f, 0.f, 0.f};
      #pragma unroll
      for (int k0 = 0; k0 < 3; ++k0) {
        v16h af = load_fragA(&sA[wave][0][0], K1PAD, m, half, k0);
        v16h bf = load_fragB(w1, K1PAD, n0 * 16 + m, half, k0);
        acc = WMMA_F16(af, bf, acc);
      }
      const int col = n0 * 16 + m;
      const float bv = bb1[col];
      #pragma unroll
      for (int r = 0; r < 8; ++r) {
        float h = acc[r] + bv; h = h > 0.f ? h : 0.f;
        sH1[wave][r + half * 8][col] = (_Float16)h;
      }
    }
    __syncthreads();

    #pragma unroll
    for (int n0 = 0; n0 < 8; ++n0) {
      v8f acc = {0.f, 0.f, 0.f, 0.f, 0.f, 0.f, 0.f, 0.f};
      #pragma unroll
      for (int k0 = 0; k0 < 4; ++k0) {
        v16h af = load_fragA(&sH1[wave][0][0], NHID, m, half, k0);
        v16h bf = load_fragB(w2, NHID, n0 * 16 + m, half, k0);
        acc = WMMA_F16(af, bf, acc);
      }
      const int col = n0 * 16 + m;
      const float bv = bb2[col];
      #pragma unroll
      for (int r = 0; r < 8; ++r) {
        float h = acc[r] + bv; h = h > 0.f ? h : 0.f;
        sH2[wave][r + half * 8][col] = (_Float16)h;
      }
    }
    __syncthreads();

    #pragma unroll
    for (int n0 = 0; n0 < 2; ++n0) {
      v8f acc = {0.f, 0.f, 0.f, 0.f, 0.f, 0.f, 0.f, 0.f};
      #pragma unroll
      for (int k0 = 0; k0 < 4; ++k0) {
        v16h af = load_fragA(&sH2[wave][0][0], NHID, m, half, k0);
        v16h bf = load_fragB(w3, NHID, n0 * 16 + m, half, k0);
        acc = WMMA_F16(af, bf, acc);
      }
      const int col = n0 * 16 + m;
      const float bv = (col < 2 * NIN) ? bb3[col] : 0.f;
      #pragma unroll
      for (int r = 0; r < 8; ++r)
        sOut[wave][r + half * 8][col] = acc[r] + bv;
    }
    __syncthreads();

    if (half == 0) {
      #pragma unroll
      for (int i = 0; i < NIN; ++i) {
        bool masked = ((i & 1) == (l & 1));
        if (!masked) {
          float sv = tanhf(sOut[wave][m][i]);
          float tv = sOut[wave][m][NIN + i];
          z[i] = z[i] * expf(sv) + tv;
          ld  += sv;
        }
      }
    }
    __syncthreads();
  }

  if (half == 0) {
    float lp = 0.f;
    #pragma unroll
    for (int i = 0; i < NIN; ++i) lp += LOG2PI_F + z[i] * z[i];
    out[rowbase + m] = ld - 0.5f * lp;
  }
}

// ---------------------------------------------------------------------------
// Host-side launch
// ---------------------------------------------------------------------------
extern "C" void kernel_launch(void* const* d_in, const int* in_sizes, int n_in,
                              void* d_out, int out_size, void* d_ws, size_t ws_size,
                              hipStream_t stream) {
  (void)in_sizes; (void)n_in; (void)out_size; (void)ws_size;

  const float* inputs  = (const float*)d_in[0];
  const float* curve   = (const float*)d_in[1];
  const float* conv1_w = (const float*)d_in[2];
  const float* conv1_b = (const float*)d_in[3];
  const float* bn1_g   = (const float*)d_in[4];
  const float* bn1_b   = (const float*)d_in[5];
  const float* rb_w1   = (const float*)d_in[6];
  const float* rb_b1   = (const float*)d_in[7];
  const float* rb_g1   = (const float*)d_in[8];
  const float* rb_be1  = (const float*)d_in[9];
  const float* rb_w2   = (const float*)d_in[10];
  const float* rb_b2   = (const float*)d_in[11];
  const float* rb_g2   = (const float*)d_in[12];
  const float* rb_be2  = (const float*)d_in[13];
  const float* lin_w   = (const float*)d_in[14];
  const float* lin_b   = (const float*)d_in[15];
  const float* W1      = (const float*)d_in[16];
  const float* b1      = (const float*)d_in[17];
  const float* W2      = (const float*)d_in[18];
  const float* b2      = (const float*)d_in[19];
  const float* W3      = (const float*)d_in[20];
  const float* b3      = (const float*)d_in[21];
  float* out = (float*)d_out;

  char* w = (char*)d_ws;
  auto carve = [&](size_t bytes) -> char* {
    char* p = w;
    w += (bytes + 255) & ~(size_t)255;
    return p;
  };
  float*    partials  = (float*)carve((size_t)BATCH * 32 * sizeof(float));
  float*    partials2 = (float*)carve((size_t)BATCH * 32 * sizeof(float));
  float*    bnp       = (float*)carve(6 * 32 * sizeof(float));
  float*    ctx       = (float*)carve((size_t)BATCH * CTXD * sizeof(float));
  _Float16* W1t       = (_Float16*)carve((size_t)NLAYERS * NHID * K1PAD * sizeof(_Float16));
  _Float16* W2t       = (_Float16*)carve((size_t)NLAYERS * NHID * NHID * sizeof(_Float16));
  _Float16* W3t       = (_Float16*)carve((size_t)NLAYERS * N3PAD * NHID * sizeof(_Float16));
  _Float16* rbB       = (_Float16*)carve((size_t)RBW_ELEMS * sizeof(_Float16));

  {
    int n1 = NLAYERS * NHID * K1PAD;
    int n2 = NLAYERS * NHID * NHID;
    int n3 = NLAYERS * N3PAD * NHID;
    prep_w1 <<<(n1 + 255) / 256, 256, 0, stream>>>(W1, W1t);
    prep_w2 <<<(n2 + 255) / 256, 256, 0, stream>>>(W2, W2t);
    prep_w3 <<<(n3 + 255) / 256, 256, 0, stream>>>(W3, W3t);
    prep_rbw<<<(RBW_ELEMS + 255) / 256, 256, 0, stream>>>(rb_w1, rb_w2, rbB);
  }

  enc_pass<<<BATCH, 256, 0, stream>>>(1, curve, conv1_w, conv1_b, rb_b1, rb_b2,
                                      rbB, lin_w, lin_b, bnp,
                                      partials, partials2, ctx);
  bn_finalize<<<32, 256, 0, stream>>>(partials, partials2, bn1_g, bn1_b, bnp + 0 * 64);

  enc_pass<<<BATCH, 256, 0, stream>>>(2, curve, conv1_w, conv1_b, rb_b1, rb_b2,
                                      rbB, lin_w, lin_b, bnp,
                                      partials, partials2, ctx);
  bn_finalize<<<32, 256, 0, stream>>>(partials, partials2, rb_g1, rb_be1, bnp + 1 * 64);

  enc_pass<<<BATCH, 256, 0, stream>>>(3, curve, conv1_w, conv1_b, rb_b1, rb_b2,
                                      rbB, lin_w, lin_b, bnp,
                                      partials, partials2, ctx);
  bn_finalize<<<32, 256, 0, stream>>>(partials, partials2, rb_g2, rb_be2, bnp + 2 * 64);

  enc_pass<<<BATCH, 256, 0, stream>>>(4, curve, conv1_w, conv1_b, rb_b1, rb_b2,
                                      rbB, lin_w, lin_b, bnp,
                                      partials, partials2, ctx);

  coupling<<<BATCH / 64, 128, 0, stream>>>(inputs, ctx, W1t, W2t, W3t,
                                           b1, b2, b3, out);
}
